// ModulatedLinear_46935402611220
// MI455X (gfx1250) — compile-verified, matrix operand order checked by reading, not run on hardware
//
#include <hip/hip_runtime.h>
#include <hip/hip_bf16.h>
#include <math.h>

// ---------------------------------------------------------------------------
// ModulatedLinear, MI455X (gfx1250), wave32 + V_WMMA_F32_16X16X32_F16.
//   s    = modulation @ mod_w^T + mod_b
//   out  = (x*s) @ W^T * rsqrt((s*s) @ (W*W)^T + eps) + bias
// Three 16384x512x512 GEMMs on the f16 WMMA pipe (f32 accumulate).
// Weight tiles staged in LDS (bank-conflict-free padding); token (A) stream
// double-buffered in registers so global-load latency overlaps the WMMAs.
// ---------------------------------------------------------------------------

typedef _Float16 v16h __attribute__((ext_vector_type(16)));
typedef _Float16 v8h  __attribute__((ext_vector_type(8)));
typedef float    v8f  __attribute__((ext_vector_type(8)));
typedef float    v4f  __attribute__((ext_vector_type(4)));

#define IN_DIM  512
#define OUT_DIM 512
#define MOD_DIM 512
#define NTOK    16384            // B*S = 4*4096
#define EPSV    1e-8f
#define LDS_LD  520              // 512 + 8 halves: row stride = 260 dwords = 4 mod 64 banks

union HFrag {
  v16h v;
  v8h  h8[2];
  _Float16 e[16];
};

// --- fragment loaders -----------------------------------------------------
// 16-bit 16x32 fragment layout (ISA 7.12.2): lanes 0-15 hold row (lane&15),
// K = [k0, k0+8) in VGPRs 0-3 and K = [k0+16, k0+24) in VGPRs 4-7; lanes
// 16-31 hold the same rows at K+8. K is contiguous for all our operands, so
// each half is one b128 load of 8 halves.
__device__ __forceinline__ v16h load_frag_f16(const _Float16* __restrict__ base,
                                              int row, int ld, int k0, int lane) {
  const int half = lane >> 4;
  const _Float16* p = base + (size_t)row * ld + k0 + half * 8;
  HFrag f;
  f.h8[0] = *reinterpret_cast<const v8h*>(p);
  f.h8[1] = *reinterpret_cast<const v8h*>(p + 16);
  return f.v;
}

// B fragment from the LDS-staged weight tile (ds_load_b128 x2, conflict-free).
__device__ __forceinline__ v16h load_frag_lds(const _Float16* base,
                                              int row, int k0, int lane) {
  const int half = lane >> 4;
  const _Float16* p = base + row * LDS_LD + k0 + half * 8;
  HFrag f;
  f.h8[0] = *reinterpret_cast<const v8h*>(p);
  f.h8[1] = *reinterpret_cast<const v8h*>(p + 16);
  return f.v;
}

// A fragment from f32 data, converted on the fly (v_cvt_pk_f16_f32).
__device__ __forceinline__ v16h load_frag_f32(const float* __restrict__ base,
                                              int row, int ld, int k0, int lane) {
  const int half = lane >> 4;
  const float* p = base + (size_t)row * ld + k0 + half * 8;
  v4f a0 = *reinterpret_cast<const v4f*>(p);
  v4f a1 = *reinterpret_cast<const v4f*>(p + 4);
  v4f b0 = *reinterpret_cast<const v4f*>(p + 16);
  v4f b1 = *reinterpret_cast<const v4f*>(p + 20);
  HFrag f;
#pragma unroll
  for (int i = 0; i < 4; ++i) {
    f.e[i]      = (_Float16)a0[i];
    f.e[4 + i]  = (_Float16)a1[i];
    f.e[8 + i]  = (_Float16)b0[i];
    f.e[12 + i] = (_Float16)b1[i];
  }
  return f.v;
}

// Cooperative stage of a 64-row x 512-col f16 tile into padded LDS.
__device__ __forceinline__ void stage_tile(_Float16* dst, const _Float16* __restrict__ src,
                                           int tid) {
  // 64 rows * 64 chunks of 8 halves = 4096 chunks; 256 threads -> 16 each.
  for (int c = tid; c < 64 * 64; c += 256) {
    const int row = c >> 6;
    const int col = (c & 63) * 8;
    *reinterpret_cast<v8h*>(&dst[row * LDS_LD + col]) =
        *reinterpret_cast<const v8h*>(src + (size_t)row * 512 + col);
  }
  __syncthreads();
}

#define WMMA_F16(A, B, C) \
  __builtin_amdgcn_wmma_f32_16x16x32_f16(false, (A), false, (B), (short)0, (C), false, false)

// --- Kernel 0: one-shot weight conversion f32 -> f16 ----------------------
__global__ __launch_bounds__(256)
void cvt_weights_kernel(const float* __restrict__ w, const float* __restrict__ mw,
                        _Float16* __restrict__ w16, _Float16* __restrict__ mw16) {
  int i = blockIdx.x * blockDim.x + threadIdx.x;
  if (i < OUT_DIM * IN_DIM) {
    w16[i]  = (_Float16)w[i];
    mw16[i] = (_Float16)mw[i];
  }
}

// --- Kernel 1: s = modulation @ mod_w^T + mod_b; emit xs = x*s, ss = s*s --
// Block: 8 waves, tile = 256 tokens x 64 features. Wave tile: 32 x 64.
__global__ __launch_bounds__(256)
void modstyle_kernel(const float* __restrict__ mod, const float* __restrict__ x,
                     const _Float16* __restrict__ mw16, const float* __restrict__ mod_b,
                     _Float16* __restrict__ xs, _Float16* __restrict__ ss) {
  __shared__ _Float16 Bs[64 * LDS_LD];
  const int lane = threadIdx.x & 31;
  const int wave = threadIdx.x >> 5;
  const int r    = lane & 15;
  const int tok0 = blockIdx.x * 256 + wave * 32;
  const int n0   = blockIdx.y * 64;

  stage_tile(Bs, mw16 + (size_t)n0 * MOD_DIM, threadIdx.x);

  v8f acc[2][4] = {};
  v16h a[2][2];  // [buffer][m-subtile]
  a[0][0] = load_frag_f32(mod, tok0 + r,      MOD_DIM, 0, lane);
  a[0][1] = load_frag_f32(mod, tok0 + 16 + r, MOD_DIM, 0, lane);

  for (int it = 0; it < MOD_DIM / 32; ++it) {
    const int cur = it & 1;
    const int k0  = it * 32;
    if (it + 1 < MOD_DIM / 32) {
      a[cur ^ 1][0] = load_frag_f32(mod, tok0 + r,      MOD_DIM, k0 + 32, lane);
      a[cur ^ 1][1] = load_frag_f32(mod, tok0 + 16 + r, MOD_DIM, k0 + 32, lane);
    }
#pragma unroll
    for (int f = 0; f < 4; ++f) {
      v16h b = load_frag_lds(Bs, f * 16 + r, k0, lane);
#pragma unroll
      for (int m = 0; m < 2; ++m)
        acc[m][f] = WMMA_F16(a[cur][m], b, acc[m][f]);
    }
  }

  // C/D layout: lane -> col (lane&15); VGPR v -> row v (lanes 0-15) or v+8.
  const int col   = lane & 15;
  const int mbase = (lane >> 4) * 8;
#pragma unroll
  for (int m = 0; m < 2; ++m) {
#pragma unroll
    for (int f = 0; f < 4; ++f) {
      const int n = n0 + f * 16 + col;
      const float bm = mod_b[n];
#pragma unroll
      for (int v = 0; v < 8; ++v) {
        const int t = tok0 + m * 16 + mbase + v;
        const float s  = acc[m][f][v] + bm;
        const float xv = x[(size_t)t * IN_DIM + n];
        xs[(size_t)t * IN_DIM + n] = (_Float16)(xv * s);
        ss[(size_t)t * IN_DIM + n] = (_Float16)(s * s);
      }
    }
  }
}

// --- Kernel 2: out = (xs @ W^T) * rsqrt(ss @ (W*W)^T + eps) + bias --------
// Dual GEMM; W^2 fragments built from the LDS W tile with packed f16 muls.
__global__ __launch_bounds__(256)
void modlinear_kernel(const _Float16* __restrict__ xs, const _Float16* __restrict__ ss,
                      const _Float16* __restrict__ w16, const float* __restrict__ bias,
                      float* __restrict__ out) {
  __shared__ _Float16 Bs[64 * LDS_LD];
  const int lane = threadIdx.x & 31;
  const int wave = threadIdx.x >> 5;
  const int r    = lane & 15;
  const int tok0 = blockIdx.x * 256 + wave * 32;
  const int n0   = blockIdx.y * 64;

  stage_tile(Bs, w16 + (size_t)n0 * IN_DIM, threadIdx.x);

  v8f accO[2][4] = {};
  v8f accD[2][4] = {};
  v16h a_xs[2][2], a_ss[2][2];  // [buffer][m-subtile]
  a_xs[0][0] = load_frag_f16(xs, tok0 + r,      IN_DIM, 0, lane);
  a_xs[0][1] = load_frag_f16(xs, tok0 + 16 + r, IN_DIM, 0, lane);
  a_ss[0][0] = load_frag_f16(ss, tok0 + r,      IN_DIM, 0, lane);
  a_ss[0][1] = load_frag_f16(ss, tok0 + 16 + r, IN_DIM, 0, lane);

  for (int it = 0; it < IN_DIM / 32; ++it) {
    const int cur = it & 1;
    const int k0  = it * 32;
    if (it + 1 < IN_DIM / 32) {
      a_xs[cur ^ 1][0] = load_frag_f16(xs, tok0 + r,      IN_DIM, k0 + 32, lane);
      a_xs[cur ^ 1][1] = load_frag_f16(xs, tok0 + 16 + r, IN_DIM, k0 + 32, lane);
      a_ss[cur ^ 1][0] = load_frag_f16(ss, tok0 + r,      IN_DIM, k0 + 32, lane);
      a_ss[cur ^ 1][1] = load_frag_f16(ss, tok0 + 16 + r, IN_DIM, k0 + 32, lane);
    }
#pragma unroll
    for (int f = 0; f < 4; ++f) {
      v16h bw  = load_frag_lds(Bs, f * 16 + r, k0, lane);
      v16h bw2 = bw * bw;  // packed f16 mul -> W^2 fragment
#pragma unroll
      for (int m = 0; m < 2; ++m) {
        accO[m][f] = WMMA_F16(a_xs[cur][m], bw,  accO[m][f]);
        accD[m][f] = WMMA_F16(a_ss[cur][m], bw2, accD[m][f]);
      }
    }
  }

  const int col   = lane & 15;
  const int mbase = (lane >> 4) * 8;
#pragma unroll
  for (int m = 0; m < 2; ++m) {
#pragma unroll
    for (int f = 0; f < 4; ++f) {
      const int n = n0 + f * 16 + col;
      const float bv = bias[n];
#pragma unroll
      for (int v = 0; v < 8; ++v) {
        const int t = tok0 + m * 16 + mbase + v;
        out[(size_t)t * OUT_DIM + n] =
            accO[m][f][v] * rsqrtf(accD[m][f][v] + EPSV) + bv;
      }
    }
  }
}

// ---------------------------------------------------------------------------
extern "C" void kernel_launch(void* const* d_in, const int* in_sizes, int n_in,
                              void* d_out, int out_size, void* d_ws, size_t ws_size,
                              hipStream_t stream) {
  // setup_inputs() order: x, modulation, weight, bias, mod_w, mod_b (all f32)
  const float* x     = (const float*)d_in[0];
  const float* mod   = (const float*)d_in[1];
  const float* w     = (const float*)d_in[2];
  const float* bias  = (const float*)d_in[3];
  const float* mod_w = (const float*)d_in[4];
  const float* mod_b = (const float*)d_in[5];
  float* out = (float*)d_out;

  // Workspace layout (33 MB):
  //   [0,      512KB) mod_w in f16
  //   [512KB,  1MB  ) W in f16
  //   [1MB,    17MB ) xs = (x*s) f16   [NTOK, IN_DIM]
  //   [17MB,   33MB ) ss = (s*s) f16   [NTOK, IN_DIM]
  char* ws = (char*)d_ws;
  _Float16* mw16 = (_Float16*)(ws);
  _Float16* w16  = (_Float16*)(ws + (512u << 10));
  _Float16* xsb  = (_Float16*)(ws + (1u << 20));
  _Float16* ssb  = (_Float16*)(ws + (1u << 20) + (16u << 20));

  (void)in_sizes; (void)n_in; (void)out_size; (void)ws_size;

  cvt_weights_kernel<<<dim3((OUT_DIM * IN_DIM + 255) / 256), dim3(256), 0, stream>>>(
      w, mod_w, w16, mw16);

  modstyle_kernel<<<dim3(NTOK / 256, IN_DIM / 64), dim3(256), 0, stream>>>(
      mod, x, mw16, mod_b, xsb, ssb);

  modlinear_kernel<<<dim3(NTOK / 256, OUT_DIM / 64), dim3(256), 0, stream>>>(
      xsb, ssb, w16, bias, out);
}
